// Attention_21715354648747
// MI455X (gfx1250) — matrix operand order repaired and checked by reading, not compiled, not run on hardware
//
#include <hip/hip_runtime.h>
#include <hip/hip_bf16.h>

#define HEAD_DIM 128
#define HIDDEN 5120
#define NUM_KV_GROUPS 8
#define NUM_MODALITIES 3
#define NUM_Q_HEADS 40
#define Q_DIM (NUM_Q_HEADS * HEAD_DIM)          /* 5120 */
#define KV_DIM (NUM_KV_GROUPS * HEAD_DIM)       /* 1024 */
#define GATE_OFF (Q_DIM + 2 * KV_DIM)           /* 7168 */
#define QKV_PER_MOD (GATE_OFF + NUM_Q_HEADS)    /* 7208 */
#define SEQ 2048

typedef __attribute__((ext_vector_type(16))) __bf16 v16bf;
typedef __attribute__((ext_vector_type(8)))  float  v8f;

union Frag16 { v16bf v; __bf16 h[16]; unsigned short u[16]; uint4 q[2]; };

__device__ __forceinline__ unsigned short f2bf(float f) {
  __bf16 h = (__bf16)f;                       // native v_cvt, RNE
  return __builtin_bit_cast(unsigned short, h);
}

__device__ __forceinline__ v8f wmma_bf16(v16bf a, v16bf b, v8f c) {
  // D = A(16x32 bf16) * B(32x16 bf16) + C(16x16 f32)
  return __builtin_amdgcn_wmma_f32_16x16x32_bf16(false, a, false, b, (short)0, c,
                                                 false, false);
}

// ---------------------------------------------------------------------------
// Kernel 0: bucket token indices by modality -> perm[], offsets[4]
// ---------------------------------------------------------------------------
__global__ void build_groups_kernel(const int* __restrict__ mod,
                                    int* __restrict__ offsets,
                                    int* __restrict__ perm) {
  __shared__ int cnt[NUM_MODALITIES];
  __shared__ int cur[NUM_MODALITIES];
  int t = threadIdx.x;
  if (t < NUM_MODALITIES) cnt[t] = 0;
  __syncthreads();
  for (int i = t; i < SEQ; i += 256) atomicAdd(&cnt[mod[i]], 1);
  __syncthreads();
  if (t == 0) {
    int off = 0;
    for (int m = 0; m < NUM_MODALITIES; ++m) {
      offsets[m] = off; cur[m] = off; off += cnt[m];
    }
    offsets[NUM_MODALITIES] = off;
  }
  __syncthreads();
  for (int i = t; i < SEQ; i += 256) {
    int pos = atomicAdd(&cur[mod[i]], 1);
    perm[pos] = i;   // grouped by modality; within-group order irrelevant
  }
}

// ---------------------------------------------------------------------------
// Kernel 1: per-token RMSNorm with modality-selected weight -> bf16 activations
// ---------------------------------------------------------------------------
__global__ __launch_bounds__(256)
void rmsnorm_kernel(const float* __restrict__ x, const int* __restrict__ mod,
                    const float* __restrict__ pre_norm,
                    unsigned short* __restrict__ xn) {
  int tok = blockIdx.x;
  const float* row = x + (size_t)tok * HIDDEN;
  const float* w   = pre_norm + (size_t)mod[tok] * HIDDEN;
  __shared__ float red[8];
  float ss = 0.f;
  #pragma unroll
  for (int j0 = 0; j0 < HIDDEN; j0 += 256) {
    float v = row[j0 + threadIdx.x]; ss += v * v;
  }
  #pragma unroll
  for (int m = 16; m >= 1; m >>= 1) ss += __shfl_xor(ss, m, 32);
  int wave = threadIdx.x >> 5;
  if ((threadIdx.x & 31) == 0) red[wave] = ss;
  __syncthreads();
  if (threadIdx.x < 8) {
    float v = red[threadIdx.x];
    #pragma unroll
    for (int m = 4; m >= 1; m >>= 1) v += __shfl_xor(v, m, 32);
    if (threadIdx.x == 0) red[0] = v;
  }
  __syncthreads();
  float inv = rsqrtf(red[0] / (float)HIDDEN + 1e-6f);
  unsigned short* out = xn + (size_t)tok * HIDDEN;
  #pragma unroll
  for (int j0 = 0; j0 < HIDDEN; j0 += 256) {
    int j = j0 + threadIdx.x;
    out[j] = f2bf(row[j] * inv * w[j]);
  }
}

// ---------------------------------------------------------------------------
// Kernel 2: grouped gather-GEMM (used for QKV and for output projection).
// C[tok, n] = sum_k A_bf16[tok, k] * W_fp32[mod, n, k]   (K = HIDDEN)
// Block: 256 thr (8 waves). Tile: 16 tokens x 512 outputs (wave = 4 subtiles).
// ---------------------------------------------------------------------------
__global__ __launch_bounds__(256)
void gemm_grouped_kernel(const unsigned short* __restrict__ Abf,
                         const float* __restrict__ W,
                         float* __restrict__ Cout,
                         const int* __restrict__ perm,
                         const int* __restrict__ offsets,
                         int N, int ldC) {
  int mz   = blockIdx.z;
  int off0 = offsets[mz], off1 = offsets[mz + 1];
  int cnt  = off1 - off0;
  int mtile = blockIdx.y * 16;
  if (mtile >= cnt) return;
  int valid = min(16, cnt - mtile);

  __shared__ int tok[16];
  __shared__ unsigned short ldsA[32 * 16];   // A fragment staged in WMMA lane layout

  int t = threadIdx.x;
  if (t < 16) tok[t] = perm[off0 + mtile + min(t, valid - 1)];

  int wave = t >> 5;
  int lane = t & 31;
  int half = lane >> 4;
  int ln   = lane & 15;
  int nBase = blockIdx.x * 512 + wave * 64;
  const float* Wm = W + (size_t)mz * N * HIDDEN;

  // Clamp out-of-range output columns: they load valid (duplicate) weight rows
  // but are never stored, so no zero-fill / branch is needed in the K loop.
  const float* wrow[4];
  #pragma unroll
  for (int s = 0; s < 4; ++s) {
    int n = min(nBase + s * 16 + ln, N - 1);
    wrow[s] = Wm + (size_t)n * HIDDEN + half * 8;
  }

  v8f acc[4];
  #pragma unroll
  for (int s = 0; s < 4; ++s)
    #pragma unroll
    for (int i = 0; i < 8; ++i) acc[s][i] = 0.f;

  int fm = t >> 3;       // token row (t<128)
  int fj = t & 7;        // 4-short chunk within 32-wide K slab
  int kr  = fj * 4;
  int blk = kr >> 3;
  int la  = fm + ((blk & 1) ? 16 : 0);
  int e   = (kr & 7) + ((blk >= 2) ? 8 : 0);
  __syncthreads();

  for (int k0 = 0; k0 < HIDDEN; k0 += 32) {
    if (t < 128) {
      const unsigned short* src = Abf + (size_t)tok[fm] * HIDDEN + k0 + kr;
      *(ushort4*)&ldsA[la * 16 + e] = *(const ushort4*)src;
    }
    __syncthreads();
    Frag16 a;
    a.q[0] = *(const uint4*)&ldsA[lane * 16];
    a.q[1] = *(const uint4*)&ldsA[lane * 16 + 8];
    #pragma unroll
    for (int s = 0; s < 4; ++s) {
      const float* wp = wrow[s] + k0;
      float4 f0 = *(const float4*)wp;
      float4 f1 = *(const float4*)(wp + 4);
      float4 f2 = *(const float4*)(wp + 16);
      float4 f3 = *(const float4*)(wp + 20);
      Frag16 b;
      b.h[0]  = (__bf16)f0.x; b.h[1]  = (__bf16)f0.y; b.h[2]  = (__bf16)f0.z; b.h[3]  = (__bf16)f0.w;
      b.h[4]  = (__bf16)f1.x; b.h[5]  = (__bf16)f1.y; b.h[6]  = (__bf16)f1.z; b.h[7]  = (__bf16)f1.w;
      b.h[8]  = (__bf16)f2.x; b.h[9]  = (__bf16)f2.y; b.h[10] = (__bf16)f2.z; b.h[11] = (__bf16)f2.w;
      b.h[12] = (__bf16)f3.x; b.h[13] = (__bf16)f3.y; b.h[14] = (__bf16)f3.z; b.h[15] = (__bf16)f3.w;
      acc[s] = wmma_bf16(a.v, b.v, acc[s]);
    }
    __syncthreads();
  }

  #pragma unroll
  for (int s = 0; s < 4; ++s) {
    int n = nBase + s * 16 + ln;
    if (n >= N) continue;
    #pragma unroll
    for (int r = 0; r < 8; ++r) {
      int mr = r + half * 8;
      if (mr < valid) Cout[(size_t)tok[mr] * ldC + n] = acc[s][r];
    }
  }
}

// ---------------------------------------------------------------------------
// Kernel 3: per-head q/k RMSNorm + RoPE; lay out Q/K/V for the attention WMMAs.
//   qbf[h][tok][d], kbf[g][tok][d] (row-major d), vt[g][d][tok] (transposed).
// ---------------------------------------------------------------------------
__global__ __launch_bounds__(256)
void qk_prep_kernel(const float* __restrict__ qkv, const int* __restrict__ mod,
                    const float* __restrict__ qnw, const float* __restrict__ knw,
                    const float* __restrict__ cosr, const float* __restrict__ sinr,
                    unsigned short* __restrict__ qbf, unsigned short* __restrict__ kbf,
                    unsigned short* __restrict__ vt, float* __restrict__ gate) {
  int tok  = blockIdx.x;
  int t    = threadIdx.x, wave = t >> 5, lane = t & 31;
  const float* row = qkv + (size_t)tok * QKV_PER_MOD;
  int mz = mod[tok];
  const float* qw = qnw + mz * HEAD_DIM;
  const float* kw = knw + mz * HEAD_DIM;
  const float* cr = cosr + (size_t)tok * HEAD_DIM;
  const float* sr = sinr + (size_t)tok * HEAD_DIM;

  // Q heads: wave w handles heads w, w+8, ... ; lane holds d = lane + 32j
  for (int h = wave; h < NUM_Q_HEADS; h += 8) {
    const float* qp = row + h * HEAD_DIM;
    float v[4]; float ss = 0.f;
    #pragma unroll
    for (int j = 0; j < 4; ++j) { v[j] = qp[lane + 32 * j]; ss += v[j] * v[j]; }
    #pragma unroll
    for (int m = 16; m >= 1; m >>= 1) ss += __shfl_xor(ss, m, 32);
    float inv = rsqrtf(ss / (float)HEAD_DIM + 1e-6f);
    float nv[4], c[4], s_[4];
    #pragma unroll
    for (int j = 0; j < 4; ++j) {
      int d = lane + 32 * j;
      nv[j] = v[j] * inv * qw[d]; c[j] = cr[d]; s_[j] = sr[d];
    }
    float o0 = nv[0] * c[0] - nv[2] * s_[0];
    float o1 = nv[1] * c[1] - nv[3] * s_[1];
    float o2 = nv[2] * c[2] + nv[0] * s_[2];
    float o3 = nv[3] * c[3] + nv[1] * s_[3];
    unsigned short* qo = qbf + ((size_t)h * SEQ + tok) * HEAD_DIM;
    qo[lane]      = f2bf(o0); qo[lane + 32] = f2bf(o1);
    qo[lane + 64] = f2bf(o2); qo[lane + 96] = f2bf(o3);
  }
  // K groups + V (waves 0..7 -> group)
  if (wave < NUM_KV_GROUPS) {
    int g = wave;
    const float* kp = row + Q_DIM + g * HEAD_DIM;
    float v[4]; float ss = 0.f;
    #pragma unroll
    for (int j = 0; j < 4; ++j) { v[j] = kp[lane + 32 * j]; ss += v[j] * v[j]; }
    #pragma unroll
    for (int m = 16; m >= 1; m >>= 1) ss += __shfl_xor(ss, m, 32);
    float inv = rsqrtf(ss / (float)HEAD_DIM + 1e-6f);
    float nv[4], c[4], s_[4];
    #pragma unroll
    for (int j = 0; j < 4; ++j) {
      int d = lane + 32 * j;
      nv[j] = v[j] * inv * kw[d]; c[j] = cr[d]; s_[j] = sr[d];
    }
    float o0 = nv[0] * c[0] - nv[2] * s_[0];
    float o1 = nv[1] * c[1] - nv[3] * s_[1];
    float o2 = nv[2] * c[2] + nv[0] * s_[2];
    float o3 = nv[3] * c[3] + nv[1] * s_[3];
    unsigned short* ko = kbf + ((size_t)g * SEQ + tok) * HEAD_DIM;
    ko[lane]      = f2bf(o0); ko[lane + 32] = f2bf(o1);
    ko[lane + 64] = f2bf(o2); ko[lane + 96] = f2bf(o3);
    const float* vp = row + Q_DIM + KV_DIM + g * HEAD_DIM;
    #pragma unroll
    for (int j = 0; j < 4; ++j) {
      int d = lane + 32 * j;
      vt[((size_t)g * HEAD_DIM + d) * SEQ + tok] = f2bf(vp[d]);
    }
  }
  if (t < NUM_Q_HEADS) {
    float gv = row[GATE_OFF + t];
    gate[(size_t)tok * NUM_Q_HEADS + t] = 1.f / (1.f + __expf(-gv));
  }
}

// ---------------------------------------------------------------------------
// Kernel 4: flash attention. 1 wave per (head, 16-query tile). Full (non-causal)
// softmax over 2048 keys in 64 chunks of 32. QK^T and PV both via bf16 WMMA.
// ---------------------------------------------------------------------------
__global__ __launch_bounds__(32)
void attention_kernel(const unsigned short* __restrict__ qbf,
                      const unsigned short* __restrict__ kbf,
                      const unsigned short* __restrict__ vt,
                      const float* __restrict__ gate,
                      unsigned short* __restrict__ attn_bf) {
  int h  = blockIdx.x;
  int q0 = blockIdx.y * 16;
  int g  = h / (NUM_Q_HEADS / NUM_KV_GROUPS);
  int lane = threadIdx.x;
  int half = lane >> 4;
  int ln   = lane & 15;

  __shared__ unsigned short ldsP[32 * 16];   // P tile C-layout -> A-layout staging

  Frag16 qf[4];
  {
    const unsigned short* qp = qbf + ((size_t)h * SEQ + q0 + ln) * HEAD_DIM;
    #pragma unroll
    for (int c = 0; c < 4; ++c) {
      int dbase = c * 32 + half * 16;
      qf[c].q[0] = *(const uint4*)(qp + dbase);
      qf[c].q[1] = *(const uint4*)(qp + dbase + 8);
    }
  }

  v8f o[8];
  #pragma unroll
  for (int s = 0; s < 8; ++s)
    #pragma unroll
    for (int i = 0; i < 8; ++i) o[s][i] = 0.f;
  float mrow[8], lrow[8];
  #pragma unroll
  for (int r = 0; r < 8; ++r) { mrow[r] = -1e30f; lrow[r] = 0.f; }
  const float scale = 0.08838834764831845f;  // 1/sqrt(128)

  for (int k0 = 0; k0 < SEQ; k0 += 32) {
    v8f sc0, sc1;
    #pragma unroll
    for (int i = 0; i < 8; ++i) { sc0[i] = 0.f; sc1[i] = 0.f; }
    #pragma unroll
    for (int c = 0; c < 4; ++c) {
      const unsigned short* kp0 =
          kbf + ((size_t)g * SEQ + k0 + ln) * HEAD_DIM + c * 32 + half * 16;
      const unsigned short* kp1 = kp0 + 16 * HEAD_DIM;
      Frag16 b0, b1;
      b0.q[0] = *(const uint4*)kp0; b0.q[1] = *(const uint4*)(kp0 + 8);
      b1.q[0] = *(const uint4*)kp1; b1.q[1] = *(const uint4*)(kp1 + 8);
      sc0 = wmma_bf16(qf[c].v, b0.v, sc0);
      sc1 = wmma_bf16(qf[c].v, b1.v, sc1);
    }
    float mnew[8], alpha[8], p0[8], p1[8];
    #pragma unroll
    for (int r = 0; r < 8; ++r) {
      float s0 = sc0[r] * scale, s1 = sc1[r] * scale;
      sc0[r] = s0; sc1[r] = s1;
      float mx = fmaxf(s0, s1);
      #pragma unroll
      for (int m = 1; m < 16; m <<= 1) mx = fmaxf(mx, __shfl_xor(mx, m, 32));
      mnew[r]  = fmaxf(mrow[r], mx);
      alpha[r] = __expf(mrow[r] - mnew[r]);
      mrow[r]  = mnew[r];
    }
    #pragma unroll
    for (int r = 0; r < 8; ++r) {
      p0[r] = __expf(sc0[r] - mnew[r]);
      p1[r] = __expf(sc1[r] - mnew[r]);
      float rs = p0[r] + p1[r];
      #pragma unroll
      for (int m = 1; m < 16; m <<= 1) rs += __shfl_xor(rs, m, 32);
      lrow[r] = lrow[r] * alpha[r] + rs;
    }
    #pragma unroll
    for (int s = 0; s < 8; ++s)
      #pragma unroll
      for (int r = 0; r < 8; ++r) o[s][r] *= alpha[r];

    __syncthreads();
    #pragma unroll
    for (int r = 0; r < 8; ++r) {      // scatter P into WMMA A-fragment layout
      int M   = r + half * 8;
      int kr0 = ln;                    // keys k0+0..15
      int kr1 = 16 + ln;               // keys k0+16..31
      int la0 = M + (((kr0 >> 3) & 1) ? 16 : 0);
      int e0  = (kr0 & 7);
      int la1 = M + (((kr1 >> 3) & 1) ? 16 : 0);
      int e1  = (kr1 & 7) + 8;
      ldsP[la0 * 16 + e0] = f2bf(p0[r]);
      ldsP[la1 * 16 + e1] = f2bf(p1[r]);
    }
    __syncthreads();
    Frag16 pf;
    pf.q[0] = *(const uint4*)&ldsP[lane * 16];
    pf.q[1] = *(const uint4*)&ldsP[lane * 16 + 8];
    #pragma unroll
    for (int s = 0; s < 8; ++s) {
      int d = s * 16 + ln;
      const unsigned short* vp = vt + ((size_t)g * HEAD_DIM + d) * SEQ + k0 + half * 8;
      Frag16 vf;
      vf.q[0] = *(const uint4*)vp; vf.q[1] = *(const uint4*)(vp + 16);
      o[s] = wmma_bf16(pf.v, vf.v, o[s]);
    }
  }

  #pragma unroll
  for (int r = 0; r < 8; ++r) {
    int tokr = q0 + r + half * 8;
    float gv = gate[(size_t)tokr * NUM_Q_HEADS + h] / lrow[r];
    #pragma unroll
    for (int s = 0; s < 8; ++s) {
      int d = s * 16 + ln;
      attn_bf[(size_t)tokr * HIDDEN + h * HEAD_DIM + d] = f2bf(o[s][r] * gv);
    }
  }
}

// ---------------------------------------------------------------------------
extern "C" void kernel_launch(void* const* d_in, const int* in_sizes, int n_in,
                              void* d_out, int out_size, void* d_ws, size_t ws_size,
                              hipStream_t stream) {
  const float* x        = (const float*)d_in[0];
  const float* rope_cos = (const float*)d_in[1];
  const float* rope_sin = (const float*)d_in[2];
  const int*   mod      = (const int*)d_in[3];
  const float* pre_norm = (const float*)d_in[4];
  const float* qkv_w    = (const float*)d_in[5];
  const float* q_norm   = (const float*)d_in[6];
  const float* k_norm   = (const float*)d_in[7];
  const float* proj_w   = (const float*)d_in[8];
  float* out = (float*)d_out;

  char* ws = (char*)d_ws;
  size_t p = 0;
  auto take = [&](size_t bytes) -> void* {
    void* r = ws + p;
    p = (p + bytes + 255) & ~(size_t)255;
    return r;
  };
  int*            perm    = (int*)take(SEQ * sizeof(int));
  int*            offsets = (int*)take((NUM_MODALITIES + 1) * sizeof(int));
  unsigned short* xn      = (unsigned short*)take((size_t)SEQ * HIDDEN * 2);
  float*          qkvbuf  = (float*)take((size_t)SEQ * QKV_PER_MOD * 4);
  unsigned short* qbf     = (unsigned short*)take((size_t)NUM_Q_HEADS * SEQ * HEAD_DIM * 2);
  unsigned short* kbf     = (unsigned short*)take((size_t)NUM_KV_GROUPS * SEQ * HEAD_DIM * 2);
  unsigned short* vt      = (unsigned short*)take((size_t)NUM_KV_GROUPS * SEQ * HEAD_DIM * 2);
  float*          gateb   = (float*)take((size_t)SEQ * NUM_Q_HEADS * 4);
  unsigned short* attnb   = (unsigned short*)take((size_t)SEQ * HIDDEN * 2);

  build_groups_kernel<<<1, 256, 0, stream>>>(mod, offsets, perm);
  rmsnorm_kernel<<<SEQ, 256, 0, stream>>>(x, mod, pre_norm, xn);

  dim3 gq((QKV_PER_MOD + 511) / 512, SEQ / 16, NUM_MODALITIES);
  gemm_grouped_kernel<<<gq, 256, 0, stream>>>(xn, qkv_w, qkvbuf, perm, offsets,
                                              QKV_PER_MOD, QKV_PER_MOD);

  qk_prep_kernel<<<SEQ, 256, 0, stream>>>(qkvbuf, mod, q_norm, k_norm,
                                          rope_cos, rope_sin, qbf, kbf, vt, gateb);

  dim3 ga(NUM_Q_HEADS, SEQ / 16);
  attention_kernel<<<ga, 32, 0, stream>>>(qbf, kbf, vt, gateb, attnb);

  dim3 gp((HIDDEN + 511) / 512, SEQ / 16, NUM_MODALITIES);
  gemm_grouped_kernel<<<gp, 256, 0, stream>>>(attnb, proj_w, out, perm, offsets,
                                              HIDDEN, HIDDEN);
}
